// AttentionWithPairBias_43044162241200
// MI455X (gfx1250) — compile-verified, matrix operand order checked by reading, not run on hardware
//
#include <hip/hip_runtime.h>

#define S_DIM 1024
#define CS 384
#define CZ 128
#define NH 12
#define HD 32

typedef __bf16 bf16;
typedef __attribute__((ext_vector_type(16))) __bf16 bf16x16;
typedef __attribute__((ext_vector_type(8)))  float  floatx8;
typedef __attribute__((ext_vector_type(4)))  unsigned int u32x4;
typedef __attribute__((ext_vector_type(4)))  float  f32x4;

union BF16Vec { bf16x16 v; u32x4 q[2]; bf16 e[16]; };

// Load 16 bf16 in the WMMA 16-bit A/B per-lane layout when the K dimension is
// contiguous in memory: elements 0..7 -> K = hi*8 + 0..7, 8..15 -> K = 16 + hi*8 + 0..7.
__device__ inline bf16x16 load_contig_tile(const bf16* base, int hi) {
  BF16Vec u;
  u.q[0] = *(const u32x4*)(base + hi * 8);
  u.q[1] = *(const u32x4*)(base + 16 + hi * 8);
  return u.v;
}

// -------------------- rmsnorm over s rows, emit bf16 --------------------
__global__ void rmsnorm_s_kernel(const float* __restrict__ x, const float* __restrict__ w,
                                 bf16* __restrict__ out) {
  __shared__ float red[128];
  int row = blockIdx.x, t = threadIdx.x;
  const float* xr = x + (size_t)row * CS;
  float v0 = xr[t], v1 = xr[t + 128], v2 = xr[t + 256];
  red[t] = v0 * v0 + v1 * v1 + v2 * v2;
  __syncthreads();
  for (int s2 = 64; s2 > 0; s2 >>= 1) {
    if (t < s2) red[t] += red[t + s2];
    __syncthreads();
  }
  float rms = rsqrtf(red[0] * (1.0f / CS) + 1e-5f);
  bf16* orow = out + (size_t)row * CS;
  orow[t]       = (bf16)(v0 * rms * w[t]);
  orow[t + 128] = (bf16)(v1 * rms * w[t + 128]);
  orow[t + 256] = (bf16)(v2 * rms * w[t + 256]);
}

// -------------------- pack f32 weight [K,N] into WMMA B-operand tile order --------------------
// Packed element index = ((kt*tiles_n + nt)*32 + lane)*16 + i
// with k = kt*32 + (i>>3)*16 + (lane>>4)*8 + (i&7), n = nt*16 + (lane&15).
__global__ void pack_b_kernel(const float* __restrict__ W, bf16* __restrict__ P, int K, int N) {
  int tid = blockIdx.x * blockDim.x + threadIdx.x;
  if (tid >= K * N) return;
  int tiles_n = N >> 4;
  int tile = tid >> 9, within = tid & 511;
  int ln = within >> 4, i = within & 15;
  int kt = tile / tiles_n, nt = tile % tiles_n;
  int hi = ln >> 4, lmn = ln & 15;
  int k = kt * 32 + ((i >> 3) << 4) + (hi << 3) + (i & 7);
  int n = nt * 16 + lmn;
  P[tid] = (bf16)W[(size_t)k * N + n];
}

// -------------------- generic wave-tile WMMA GEMM: C = A[M,K=384] @ B[384,N] --------------------
// MODE 0: store bf16; MODE 1: store f32 + bias[n]; MODE 2: f32 (x+bias[n])*gmul[m,n];
// MODE 3: store bf16 transposed (Ob[n*M + m])
template <int MODE>
__global__ void gemm16(const bf16* __restrict__ A, const bf16* __restrict__ Bp,
                       int M, int N,
                       float* __restrict__ Of, bf16* __restrict__ Ob,
                       const float* __restrict__ bias, const float* __restrict__ gmul) {
  int wid = (int)((blockIdx.x * blockDim.x + threadIdx.x) >> 5);
  int lane = threadIdx.x & 31;
  int hi = lane >> 4, lm = lane & 15;
  int tiles_n = N >> 4;
  int tn = wid % tiles_n, tm = wid / tiles_n;
  if (tm * 16 >= M) return;  // wave-uniform; grid sized exactly in practice

  const bf16* arow = A + (size_t)(tm * 16 + lm) * CS;
  floatx8 acc = {0, 0, 0, 0, 0, 0, 0, 0};
#pragma unroll
  for (int ks = 0; ks < CS / 32; ++ks) {
    bf16x16 a = load_contig_tile(arow + ks * 32, hi);
    BF16Vec b;
    const bf16* bp = Bp + (((size_t)(ks * tiles_n + tn)) << 9) + lane * 16;
    b.q[0] = *(const u32x4*)bp;
    b.q[1] = *(const u32x4*)(bp + 8);
    acc = __builtin_amdgcn_wmma_f32_16x16x32_bf16(false, a, false, b.v, (short)0, acc,
                                                  false, false);
  }
  int col = tn * 16 + lm;
#pragma unroll
  for (int r = 0; r < 8; ++r) {
    int row = tm * 16 + r + hi * 8;  // C-layout: lanes 16..31 hold rows m+8
    if (MODE == 0)      Ob[(size_t)row * N + col] = (bf16)acc[r];
    else if (MODE == 1) Of[(size_t)row * N + col] = acc[r] + bias[col];
    else if (MODE == 2) Of[(size_t)row * N + col] =
        (acc[r] + bias[col]) * gmul[(size_t)row * N + col];
    else                Ob[(size_t)col * M + row] = (bf16)acc[r];  // transposed
  }
}

// -------------------- fused pair bias: rmsnorm(z) @ Wz + mask --------------------
// One THREAD per (i,j) pair: no cross-lane ops; z row read twice (2nd pass hits WGP$),
// Wz / w_z indices are wave-uniform -> scalar loads; stores lane-coalesced per head.
__global__ void pair_bias_kernel(const float* __restrict__ z, const int* __restrict__ zmask,
                                 const float* __restrict__ w_z, const float* __restrict__ Wz,
                                 float* __restrict__ bias) {
  size_t p = (size_t)blockIdx.x * blockDim.x + threadIdx.x;
  const float* zr = z + p * CZ;
  // pass 1: sum of squares
  float sq = 0.0f;
#pragma unroll
  for (int c = 0; c < CZ / 4; ++c) {
    f32x4 zv = *(const f32x4*)(zr + c * 4);
    sq += zv[0] * zv[0] + zv[1] * zv[1] + zv[2] * zv[2] + zv[3] * zv[3];
  }
  float rms = rsqrtf(sq * (1.0f / CZ) + 1e-5f);
  // pass 2: projected bias per head
  float acc[NH];
#pragma unroll
  for (int h = 0; h < NH; ++h) acc[h] = 0.0f;
#pragma unroll
  for (int c = 0; c < CZ / 4; ++c) {
    f32x4 zv = *(const f32x4*)(zr + c * 4);
#pragma unroll
    for (int t = 0; t < 4; ++t) {
      float zn = zv[t] * rms * w_z[c * 4 + t];
      const float* wrow = Wz + (c * 4 + t) * NH;
#pragma unroll
      for (int h = 0; h < NH; ++h) acc[h] += zn * wrow[h];
    }
  }
  float mv = zmask[p] ? 0.0f : -1e9f;
#pragma unroll
  for (int h = 0; h < NH; ++h)
    bias[(size_t)h * S_DIM * S_DIM + p] = acc[h] + mv;
}

// -------------------- flash attention: one wave per (head, 16-query tile) --------------------
// Q,K: [S, H*D] bf16; V: transposed [H*D, S] bf16 so the PV B-operand loads contiguously.
__global__ void attn_kernel(const bf16* __restrict__ Q, const bf16* __restrict__ Kb,
                            const bf16* __restrict__ Vt, const float* __restrict__ bias,
                            bf16* __restrict__ O) {
  __shared__ __align__(16) bf16 pstage[4][16 * 32];
  int wblk = threadIdx.x >> 5;
  int wid = blockIdx.x * 4 + wblk;
  int lane = threadIdx.x & 31, hi = lane >> 4, lm = lane & 15;
  int h = wid >> 6;   // 64 query tiles per head
  int qt = wid & 63;
  int i0 = qt << 4;
  const float scale = 0.17677669529663687f;  // 1/sqrt(32)

  bf16x16 a_q = load_contig_tile(Q + (size_t)(i0 + lm) * CS + h * HD, hi);

  floatx8 acc0 = {0, 0, 0, 0, 0, 0, 0, 0};
  floatx8 acc1 = {0, 0, 0, 0, 0, 0, 0, 0};
  float rmax[8], rsum[8];
#pragma unroll
  for (int r = 0; r < 8; ++r) { rmax[r] = -1e30f; rsum[r] = 0.0f; }

  bf16* lds = pstage[wblk];
  const floatx8 zc = {0, 0, 0, 0, 0, 0, 0, 0};
  const bf16* vb0_row = Vt + (size_t)(h * HD + lm) * S_DIM;
  const bf16* vb1_row = Vt + (size_t)(h * HD + 16 + lm) * S_DIM;

  for (int c = 0; c < S_DIM / 32; ++c) {
    int j0 = c * 32;
    // prefetch next chunk's bias lines (covers 16 rows via lm, both halves via hi)
    if (c + 1 < S_DIM / 32)
      __builtin_prefetch(bias + ((size_t)h * S_DIM + (i0 + lm)) * S_DIM + j0 + 32 + hi * 16,
                         0, 1);
    // K^T tiles: lane = key column j, K-dim = d contiguous in memory
    bf16x16 kb0 = load_contig_tile(Kb + (size_t)(j0 + lm) * CS + h * HD, hi);
    bf16x16 kb1 = load_contig_tile(Kb + (size_t)(j0 + 16 + lm) * CS + h * HD, hi);
    floatx8 s0 = __builtin_amdgcn_wmma_f32_16x16x32_bf16(false, a_q, false, kb0, (short)0,
                                                         zc, false, false);
    floatx8 s1 = __builtin_amdgcn_wmma_f32_16x16x32_bf16(false, a_q, false, kb1, (short)0,
                                                         zc, false, false);
#pragma unroll
    for (int r = 0; r < 8; ++r) {
      int m = r + hi * 8;
      const float* bp = bias + ((size_t)h * S_DIM + (i0 + m)) * S_DIM + j0;
      float v0 = s0[r] * scale + bp[lm];
      float v1 = s1[r] * scale + bp[16 + lm];
      float t = fmaxf(v0, v1);
#pragma unroll
      for (int off = 8; off >= 1; off >>= 1) t = fmaxf(t, __shfl_xor(t, off, 32));
      float nm = fmaxf(rmax[r], t);
      float corr = __expf(rmax[r] - nm);
      float p0 = __expf(v0 - nm), p1 = __expf(v1 - nm);
      float ps = p0 + p1;
#pragma unroll
      for (int off = 8; off >= 1; off >>= 1) ps += __shfl_xor(ps, off, 32);
      rsum[r] = rsum[r] * corr + ps;
      rmax[r] = nm;
      acc0[r] *= corr;
      acc1[r] *= corr;
      // stage P (C-layout) into LDS for A-layout re-read
      lds[m * 32 + lm] = (bf16)p0;
      lds[m * 32 + 16 + lm] = (bf16)p1;
    }
    asm volatile("s_wait_dscnt 0" ::: "memory");  // same-wave LDS ordering: stores done
    BF16Vec pa;
    const bf16* lp = lds + lm * 32 + hi * 8;
    pa.q[0] = *(const u32x4*)lp;
    pa.q[1] = *(const u32x4*)(lp + 16);
    // V tiles from transposed V: K=j contiguous per lane, two b128 loads each
    bf16x16 vb0 = load_contig_tile(vb0_row + j0, hi);
    bf16x16 vb1 = load_contig_tile(vb1_row + j0, hi);
    acc0 = __builtin_amdgcn_wmma_f32_16x16x32_bf16(false, pa.v, false, vb0, (short)0,
                                                   acc0, false, false);
    acc1 = __builtin_amdgcn_wmma_f32_16x16x32_bf16(false, pa.v, false, vb1, (short)0,
                                                   acc1, false, false);
  }
#pragma unroll
  for (int r = 0; r < 8; ++r) {
    float inv = 1.0f / rsum[r];
    int row = i0 + r + hi * 8;
    bf16* orow = O + (size_t)row * CS + h * HD;
    orow[lm] = (bf16)(acc0[r] * inv);
    orow[16 + lm] = (bf16)(acc1[r] * inv);
  }
}

extern "C" void kernel_launch(void* const* d_in, const int* in_sizes, int n_in,
                              void* d_out, int out_size, void* d_ws, size_t ws_size,
                              hipStream_t stream) {
  const float* s   = (const float*)d_in[0];
  const float* z   = (const float*)d_in[1];
  const int*   zm  = (const int*)d_in[2];
  const float* w_s = (const float*)d_in[3];
  const float* w_z = (const float*)d_in[4];
  const float* Wz  = (const float*)d_in[5];
  const float* Wq  = (const float*)d_in[6];
  const float* Wk  = (const float*)d_in[7];
  const float* Wv  = (const float*)d_in[8];
  const float* Wg  = (const float*)d_in[9];
  const float* bg  = (const float*)d_in[10];
  const float* Wo  = (const float*)d_in[11];
  const float* bo  = (const float*)d_in[12];
  float* out = (float*)d_out;

  char* p = (char*)d_ws;
  auto alloc = [&](size_t bytes) {
    char* r = p;
    p += (bytes + 255) & ~(size_t)255;
    return r;
  };
  bf16* s_bf = (bf16*)alloc((size_t)S_DIM * CS * 2);
  bf16* wq_p = (bf16*)alloc((size_t)CS * CS * 2);
  bf16* wk_p = (bf16*)alloc((size_t)CS * CS * 2);
  bf16* wv_p = (bf16*)alloc((size_t)CS * CS * 2);
  bf16* wg_p = (bf16*)alloc((size_t)CS * CS * 2);
  bf16* wo_p = (bf16*)alloc((size_t)CS * CS * 2);
  bf16* q_bf = (bf16*)alloc((size_t)S_DIM * CS * 2);
  bf16* k_bf = (bf16*)alloc((size_t)S_DIM * CS * 2);
  bf16* vT_bf = (bf16*)alloc((size_t)CS * S_DIM * 2);  // transposed V
  bf16* o_bf = (bf16*)alloc((size_t)S_DIM * CS * 2);
  float* g_f = (float*)alloc((size_t)S_DIM * CS * 4);
  float* bias = (float*)alloc((size_t)NH * S_DIM * S_DIM * 4);

  rmsnorm_s_kernel<<<S_DIM, 128, 0, stream>>>(s, w_s, s_bf);

  int packBlocks = (CS * CS + 255) / 256;
  pack_b_kernel<<<packBlocks, 256, 0, stream>>>(Wq, wq_p, CS, CS);
  pack_b_kernel<<<packBlocks, 256, 0, stream>>>(Wk, wk_p, CS, CS);
  pack_b_kernel<<<packBlocks, 256, 0, stream>>>(Wv, wv_p, CS, CS);
  pack_b_kernel<<<packBlocks, 256, 0, stream>>>(Wg, wg_p, CS, CS);
  pack_b_kernel<<<packBlocks, 256, 0, stream>>>(Wo, wo_p, CS, CS);

  int gemmBlocks = (S_DIM / 16) * (CS / 16) / 4;  // 4 waves per block
  gemm16<0><<<gemmBlocks, 128, 0, stream>>>(s_bf, wq_p, S_DIM, CS, nullptr, q_bf,
                                            nullptr, nullptr);
  gemm16<0><<<gemmBlocks, 128, 0, stream>>>(s_bf, wk_p, S_DIM, CS, nullptr, k_bf,
                                            nullptr, nullptr);
  gemm16<3><<<gemmBlocks, 128, 0, stream>>>(s_bf, wv_p, S_DIM, CS, nullptr, vT_bf,
                                            nullptr, nullptr);
  gemm16<1><<<gemmBlocks, 128, 0, stream>>>(s_bf, wg_p, S_DIM, CS, g_f, nullptr, bg,
                                            nullptr);

  pair_bias_kernel<<<(S_DIM * S_DIM) / 256, 256, 0, stream>>>(z, zm, w_z, Wz, bias);

  attn_kernel<<<NH * (S_DIM / 16) / 4, 128, 0, stream>>>(q_bf, k_bf, vT_bf, bias, o_bf);

  gemm16<2><<<gemmBlocks, 128, 0, stream>>>(o_bf, wo_p, S_DIM, CS, out, nullptr, bo,
                                            g_f);
}
